// Co_Spatial_Multihead_Self_Attention_14190571946231
// MI455X (gfx1250) — matrix operand order repaired and checked by reading, not compiled, not run on hardware
//
#include <hip/hip_runtime.h>
#include <hip/hip_bf16.h>

typedef unsigned short u16;
typedef unsigned int   u32;
typedef __attribute__((ext_vector_type(16))) __bf16 bf16x16;
typedef __attribute__((ext_vector_type(8)))  float  floatx8;

#define DIMC   224
#define INNERC 224
#define NHEADS 8
#define DHC    28
#define SEQL   64
#define AGCC   112

// ---- bf16 helpers (RTNE pack, bit-level; no __bf16 arithmetic needed) ----
__device__ __forceinline__ u16 f2bf(float f) {
  u32 u = __float_as_uint(f);
  u32 r = (u + 0x7FFFu + ((u >> 16) & 1u)) >> 16;
  return (u16)r;
}

union FragU { uint4 u4[2]; bf16x16 v; };

// A-fragment (16x32 bf16, M x K): lane m=lane&15, half=lane>>4.
// elements 0..7  <- K = kb + 8*half + 0..7
// elements 8..15 <- K = kb + 16 + 8*half + 0..7
__device__ __forceinline__ bf16x16 load_afrag(const u16* base, int stride,
                                              int m0, int kb, int lane) {
  int m = lane & 15, half = lane >> 4;
  const u16* p = base + (size_t)(m0 + m) * stride + kb + 8 * half;
  FragU f;
  f.u4[0] = *(const uint4*)(p);
  f.u4[1] = *(const uint4*)(p + 16);
  return f.v;
}

// B-fragment (32x16 bf16, K x N) from transposed [N][K] storage:
// lane n=lane&15, half=lane>>4; elements 0..7 <- K = kb + 8*half + e,
// elements 8..15 <- K = kb + 16 + 8*half + (e-8)
__device__ __forceinline__ bf16x16 load_bfrag(const u16* base, int stride,
                                              int n0, int kb, int lane) {
  int n = lane & 15, half = lane >> 4;
  const u16* p = base + (size_t)(n0 + n) * stride + kb + 8 * half;
  FragU f;
  f.u4[0] = *(const uint4*)(p);
  f.u4[1] = *(const uint4*)(p + 16);
  return f.v;
}

__device__ __forceinline__ floatx8 wmma_bf16(bf16x16 a, bf16x16 b, floatx8 c) {
#if defined(__gfx1250__)
  return __builtin_amdgcn_wmma_f32_16x16x32_bf16(false, a, false, b,
                                                 (short)0, c, false, false);
#else
  return c;  // host-pass stub
#endif
}

// ---------------------------------------------------------------------------
// Kernel 0: weight prep -> bf16 transposed operands in workspace
//   Wvt[np<256][k<224] = (d<28) ? Wv[k][28h+d] : 0   (np = 32h+d, head-padded)
//   Wot[n<224][k<224]  = Wo[k][n]
// ---------------------------------------------------------------------------
__global__ void wprep_kernel(const float* __restrict__ Wv,
                             const float* __restrict__ Wo,
                             u16* __restrict__ Wvt, u16* __restrict__ Wot) {
  int idx = blockIdx.x * blockDim.x + threadIdx.x;
  if (idx < 256 * DIMC) {
    int np = idx / DIMC, k = idx % DIMC;
    int h = np >> 5, d = np & 31;
    float v = (d < DHC) ? Wv[(size_t)k * INNERC + h * DHC + d] : 0.0f;
    Wvt[idx] = f2bf(v);
  }
  if (idx < DIMC * DIMC) {
    int n = idx / DIMC, k = idx % DIMC;
    Wot[idx] = f2bf(Wo[(size_t)k * DIMC + n]);
  }
}

// ---------------------------------------------------------------------------
// Kernel 1a: adaptive avg pool y: [4,256,256,112] -> py[4,64,112] (32x32 mean)
// ---------------------------------------------------------------------------
__global__ void pool_y_kernel(const float* __restrict__ y, float* __restrict__ py) {
  int blk = blockIdx.x;            // (b*64 + ci*8 + cj)
  int b = blk >> 6, cell = blk & 63, ci = cell >> 3, cj = cell & 7;
  int c = threadIdx.x;
  if (c >= AGCC) return;
  const float* base =
      y + (((size_t)(b * 256 + ci * 32)) * 256 + cj * 32) * AGCC + c;
  float s = 0.0f;
  for (int p = 0; p < 32; ++p) {
    const float* rp = base + (size_t)p * 256 * AGCC;
#pragma unroll 4
    for (int q = 0; q < 32; ++q) s += rp[(size_t)q * AGCC];
  }
  py[(size_t)blk * AGCC + c] = s * (1.0f / 1024.0f);
}

// ---------------------------------------------------------------------------
// Kernel 1b: Kmat[b*64+j][224] = (py_row @ W_ag) @ Wk     (one row per block)
// ---------------------------------------------------------------------------
__global__ void kproj_kernel(const float* __restrict__ Wag,
                             const float* __restrict__ Wk,
                             const float* __restrict__ py,
                             float* __restrict__ Kmat) {
  __shared__ float pyrow[AGCC];
  __shared__ float gkrow[DIMC];
  int row = blockIdx.x, t = threadIdx.x;
  if (t < AGCC) pyrow[t] = py[(size_t)row * AGCC + t];
  __syncthreads();
  if (t < DIMC) {
    float s = 0.0f;
    for (int k = 0; k < AGCC; ++k) s += pyrow[k] * Wag[(size_t)k * DIMC + t];
    gkrow[t] = s;
  }
  __syncthreads();
  if (t < INNERC) {
    float s = 0.0f;
    for (int k = 0; k < DIMC; ++k) s += gkrow[k] * Wk[(size_t)k * INNERC + t];
    Kmat[(size_t)row * INNERC + t] = s;
  }
}

// ---------------------------------------------------------------------------
// Kernel 1c: Mt[(b*8+h)*64+j][c<224] = bf16( scale * sum_d Wq[c][28h+d]*K[b,j,28h+d] )
//   This is the transposed B-operand for sim = X @ (scale * Wq_h K_h^T).
// ---------------------------------------------------------------------------
__global__ void mt_kernel(const float* __restrict__ Wq,
                          const float* __restrict__ Kmat,
                          const float* __restrict__ scale,
                          u16* __restrict__ Mt) {
  __shared__ float kv[DHC];
  int blk = blockIdx.x;            // (b*8+h)*64 + j
  int j = blk & 63, bh = blk >> 6, h = bh & 7, b = bh >> 3;
  int t = threadIdx.x;
  if (t < DHC) kv[t] = Kmat[((size_t)(b * 64 + j)) * INNERC + h * DHC + t];
  __syncthreads();
  if (t < DIMC) {
    float s = 0.0f;
#pragma unroll
    for (int d = 0; d < DHC; ++d)
      s += Wq[(size_t)t * INNERC + h * DHC + d] * kv[d];
    Mt[(size_t)blk * DIMC + t] = f2bf(s * scale[0]);
  }
}

// ---------------------------------------------------------------------------
// Kernel 2: fused per-window attention. One block = one 8x8 window (4096).
// 256 threads = 8 waves; waves = heads in the attention stages.
// LDS: Xs[64][232] bf16 (X, later reused as O), Vt[256][64] bf16 (V^T, padded
// heads), As[64][64] bf16 (attn probs).  69 KB dynamic.
// ---------------------------------------------------------------------------
__global__ void __launch_bounds__(256)
attn_kernel(const float* __restrict__ x, const float* __restrict__ pe,
            const float* __restrict__ bo, const u16* __restrict__ Mt,
            const u16* __restrict__ Wvt, const u16* __restrict__ Wot,
            float* __restrict__ out) {
  extern __shared__ u16 smem[];
  u16* Xs = smem;                    // 64 x 232 (stride keeps 16B alignment)
  u16* Vt = Xs + 64 * 232;           // 256 x 64
  u16* As = Vt + 256 * 64;           // 64 x 64

  const int f = blockIdx.x;
  const int b0 = f >> 10, ih = (f >> 5) & 31, iw = f & 31;
  const int kbatch = f & 3;          // reference's reshape(cwin,b,...) quirk
  const int tid = threadIdx.x, wid = tid >> 5, lane = tid & 31;
  const int mrow = lane & 15, half = lane >> 4;

  // ---- Stage 0: load x window -> Xs (f32 -> bf16) --------------------------
  {
    int t = tid >> 2, part = tid & 3;
    const float* xb =
        x + (((size_t)(b0 * 256 + ih * 8 + (t >> 3))) * 256 + iw * 8 + (t & 7)) *
                DIMC + part * 56;
    u16* dst = Xs + t * 232 + part * 56;
#pragma unroll
    for (int k4 = 0; k4 < 14; ++k4) {
      float4 v = *(const float4*)(xb + k4 * 4);
      u32 p0 = (u32)f2bf(v.x) | ((u32)f2bf(v.y) << 16);
      u32 p1 = (u32)f2bf(v.z) | ((u32)f2bf(v.w) << 16);
      *(uint2*)(dst + k4 * 4) = make_uint2(p0, p1);
    }
  }
  __syncthreads();

  // ---- Stage A: V = X @ Wv (head-padded N=256), write V^T into Vt ----------
  {
#pragma unroll
    for (int nn = 0; nn < 2; ++nn) {
      int n0 = 16 * (2 * wid + nn);
      floatx8 acc[4] = {{}, {}, {}, {}};
      for (int kb = 0; kb < DIMC; kb += 32) {
        bf16x16 bf = load_bfrag(Wvt, DIMC, n0, kb, lane);
#pragma unroll
        for (int mt = 0; mt < 4; ++mt) {
          bf16x16 af = load_afrag(Xs, 232, 16 * mt, kb, lane);
          acc[mt] = wmma_bf16(af, bf, acc[mt]);
        }
      }
#pragma unroll
      for (int mt = 0; mt < 4; ++mt) {
        // D elem r = V[16mt+8*half+r][n0+mrow] -> Vt row n0+mrow, 8 contiguous
        u32 pk[4];
#pragma unroll
        for (int p = 0; p < 4; ++p)
          pk[p] = (u32)f2bf(acc[mt][2 * p]) | ((u32)f2bf(acc[mt][2 * p + 1]) << 16);
        *(uint4*)&Vt[(size_t)(n0 + mrow) * 64 + 16 * mt + 8 * half] =
            make_uint4(pk[0], pk[1], pk[2], pk[3]);
      }
    }
  }

  // ---- Stage B: sim = X @ Mt[head] + pos ; softmax ; As = attn (bf16) ------
  {
    const int h = wid;
    const u16* Mtb = Mt + ((size_t)(kbatch * NHEADS + h)) * SEQL * DIMC;
    const float* peh = pe + (size_t)h * SEQL * SEQL;
    for (int it = 0; it < 4; ++it) {
      floatx8 acc[4] = {{}, {}, {}, {}};
      for (int kb = 0; kb < DIMC; kb += 32) {
        bf16x16 af = load_afrag(Xs, 232, 16 * it, kb, lane);
#pragma unroll
        for (int jt = 0; jt < 4; ++jt) {
          bf16x16 bf = load_bfrag(Mtb, DIMC, 16 * jt, kb, lane);
          acc[jt] = wmma_bf16(af, bf, acc[jt]);
        }
      }
#pragma unroll
      for (int jt = 0; jt < 4; ++jt)
#pragma unroll
        for (int r = 0; r < 8; ++r)
          acc[jt][r] += peh[(size_t)(16 * it + 8 * half + r) * SEQL + 16 * jt + mrow];
#pragma unroll
      for (int r = 0; r < 8; ++r) {
        float mx = fmaxf(fmaxf(acc[0][r], acc[1][r]), fmaxf(acc[2][r], acc[3][r]));
        for (int off = 1; off < 16; off <<= 1)
          mx = fmaxf(mx, __shfl_xor(mx, off, 32));   // stays within 16-lane half
        float e0 = __expf(acc[0][r] - mx), e1 = __expf(acc[1][r] - mx);
        float e2 = __expf(acc[2][r] - mx), e3 = __expf(acc[3][r] - mx);
        float s = e0 + e1 + e2 + e3;
        for (int off = 1; off < 16; off <<= 1) s += __shfl_xor(s, off, 32);
        float inv = 1.0f / s;
        u16* arow = As + (size_t)(16 * it + 8 * half + r) * 64 + mrow;
        arow[0]  = f2bf(e0 * inv);
        arow[16] = f2bf(e1 * inv);
        arow[32] = f2bf(e2 * inv);
        arow[48] = f2bf(e3 * inv);
      }
    }
  }
  __syncthreads();

  // ---- Stage C: O = attn @ V (per head); write compact O into Xs buffer ----
  {
    const int h = wid;
#pragma unroll
    for (int it = 0; it < 4; ++it) {
      bf16x16 a0 = load_afrag(As, 64, 16 * it, 0, lane);
      bf16x16 a1 = load_afrag(As, 64, 16 * it, 32, lane);
#pragma unroll
      for (int nt = 0; nt < 2; ++nt) {
        floatx8 acc = {};
        bf16x16 b0 = load_bfrag(Vt, 64, 32 * h + 16 * nt, 0, lane);
        bf16x16 b1 = load_bfrag(Vt, 64, 32 * h + 16 * nt, 32, lane);
        acc = wmma_bf16(a0, b0, acc);
        acc = wmma_bf16(a1, b1, acc);
        int dd = 16 * nt + mrow;                 // channel within head (padded)
        if (dd < DHC) {
#pragma unroll
          for (int r = 0; r < 8; ++r)
            Xs[(size_t)(16 * it + 8 * half + r) * 232 + DHC * h + dd] =
                f2bf(acc[r]);
        }
      }
    }
  }
  __syncthreads();

  // ---- Stage D: out = O @ Wo + bo, scatter to output window ----------------
  {
#pragma unroll
    for (int q = 0; q < 7; ++q) {
      int tau = wid * 7 + q;                     // 56 tiles: 4 m x 14 n
      int mt = tau & 3, nt = tau >> 2;
      floatx8 acc = {};
      for (int kb = 0; kb < INNERC; kb += 32) {
        bf16x16 af = load_afrag(Xs, 232, 16 * mt, kb, lane);
        bf16x16 bf = load_bfrag(Wot, DIMC, 16 * nt, kb, lane);
        acc = wmma_bf16(af, bf, acc);
      }
      float bias = bo[16 * nt + mrow];
#pragma unroll
      for (int r = 0; r < 8; ++r) {
        int i = 16 * mt + 8 * half + r;          // token in window
        size_t o =
            (((size_t)(b0 * 256 + ih * 8 + (i >> 3))) * 256 + iw * 8 + (i & 7)) *
                DIMC + 16 * nt + mrow;
        out[o] = acc[r] + bias;
      }
    }
  }
}

// ---------------------------------------------------------------------------
extern "C" void kernel_launch(void* const* d_in, const int* in_sizes, int n_in,
                              void* d_out, int out_size, void* d_ws, size_t ws_size,
                              hipStream_t stream) {
  const float* x     = (const float*)d_in[0];
  const float* y     = (const float*)d_in[1];
  const float* Wag   = (const float*)d_in[2];
  const float* Wq    = (const float*)d_in[3];
  const float* Wk    = (const float*)d_in[4];
  const float* Wv    = (const float*)d_in[5];
  const float* Wo    = (const float*)d_in[6];
  const float* bo    = (const float*)d_in[7];
  const float* pe    = (const float*)d_in[8];
  const float* scale = (const float*)d_in[9];
  float* out = (float*)d_out;

  char* ws = (char*)d_ws;                  // all offsets 256B-aligned
  float* py   = (float*)(ws + 0);          //   4*64*112  f32 = 114688 B
  float* Kmat = (float*)(ws + 114688);     //   4*64*224  f32 = 229376 B
  u16*   Mt   = (u16*)  (ws + 344064);     // 4*8*64*224 bf16 = 917504 B
  u16*   Wvt  = (u16*)  (ws + 1261568);    //    256*224 bf16 = 114688 B
  u16*   Wot  = (u16*)  (ws + 1376256);    //    224*224 bf16 = 100352 B

  wprep_kernel <<<224, 256, 0, stream>>>(Wv, Wo, Wvt, Wot);
  pool_y_kernel<<<256, 128, 0, stream>>>(y, py);
  kproj_kernel <<<256, 256, 0, stream>>>(Wag, Wk, py, Kmat);
  mt_kernel    <<<2048, 256, 0, stream>>>(Wq, Kmat, scale, Mt);

  size_t smem_bytes = (size_t)(64 * 232 + 256 * 64 + 64 * 64) * sizeof(u16); // 70656
  attn_kernel<<<4096, 256, smem_bytes, stream>>>(x, pe, bo, Mt, Wvt, Wot, out);
}
